// BiCLSTM_block_53360673685818
// MI455X (gfx1250) — matrix operand order repaired and checked by reading, not compiled, not run on hardware
//
#include <hip/hip_runtime.h>

typedef __attribute__((ext_vector_type(16))) _Float16 v16h;
typedef __attribute__((ext_vector_type(8)))  float    v8f;

#define S_    16
#define B_    4
#define CIN   32
#define CO    32
#define HH    128
#define WW    128
#define KDIM  64            // CIN + CO
#define M4    128           // 4*CO gate channels
#define NPX   16            // pixels per tile (along W)
#define HALO  18            // 16 + 2 halo columns
#define APACK_HALVES 73728  // 8 mtiles * 9 taps * 2 kchunks * 32 lanes * 16 halves

// ---------------------------------------------------------------------------
// Repack f32 weights [4*Co][Cin+Co][3][3] into this code's own per-lane f16 A-matrix
// fragment layout (contiguous k) fed to V_WMMA_F32_16X16X32_F16 (16x32 tile per tap/chunk), so the
// step kernel fetches each A fragment with a single 32B vector load.
//   half index = (((mt*9 + tap)*2 + kc)*32 + lane)*16 + j
//   lanes 0-15 : K=j      (K = 0..15)
//   lanes16-31 : K=j+16   (K = 16..31)
// ---------------------------------------------------------------------------
__global__ __launch_bounds__(256) void repack_weights(
    const float* __restrict__ Wf, const float* __restrict__ Wb,
    _Float16* __restrict__ apack)
{
  int gid = blockIdx.x * blockDim.x + threadIdx.x;
  if (gid >= 2 * APACK_HALVES) return;
  int dir = gid / APACK_HALVES;
  int rem = gid % APACK_HALVES;
  int j    = rem & 15;
  int lane = (rem >> 4) & 31;
  int kc   = (rem >> 9) & 1;
  int tap  = (rem >> 10) % 9;
  int mt   = rem / 9216;
  int kin  = (lane < 16) ? j
                         : j + 16;
  int o = mt * 16 + (lane & 15);
  int c = kc * 32 + kin;
  const float* W = dir ? Wb : Wf;
  apack[gid] = (_Float16)W[(o * KDIM + c) * 9 + tap];
}

// ---------------------------------------------------------------------------
// One ConvLSTM timestep for both directions (blockIdx.y = dir).
// Block = 256 threads = 8 wave32. Block tile: 16 pixels (one W-row segment) x
// all 128 gate channels; wave w owns gate channels [16w, 16w+16).
// Implicit GEMM: M=128, N=16, K=64ch*9taps, via 18 wmma_f32_16x16x32_f16.
// ---------------------------------------------------------------------------
__global__ __launch_bounds__(256) void convlstm_step(
    const float*    __restrict__ x,      // [S,B,CIN,H,W]
    const _Float16* __restrict__ apack,  // [2][APACK_HALVES]
    const float*    __restrict__ bf,     // [128]
    const float*    __restrict__ bb,     // [128]
    float*          __restrict__ out,    // [S,B,2*CO,H,W]
    float*          __restrict__ cstate, // [2][B,CO,H,W]
    int s)
{
  __shared__ __align__(32) _Float16 lin[3 * HALO * KDIM]; // 6912 B input halo (f16)
  __shared__ float zl[M4 * NPX];                          // 8192 B gate tile (f32)

  const int tid = threadIdx.x;
  const int dir = blockIdx.y;
  const int pt  = blockIdx.x;
  const int x0  = (pt & 7) << 4;
  const int y   = (pt >> 3) & 127;
  const int b   = pt >> 10;

  const int s_in = dir ? (S_ - 1 - s) : s;
  const float* xt    = x   + (size_t)(s_in * B_ + b) * CIN * (HH * WW);
  const float* hprev = out + ((size_t)((s - 1) * B_ + b) * (2 * CO) + dir * CO) * (HH * WW);

  // ---- stage input halo (3 rows x 18 cols x 64 ch) into LDS as f16 ----
  for (int e = tid; e < KDIM * 54; e += 256) {
    int ch  = e / 54;
    int pos = e % 54;
    int r = pos / HALO, q = pos % HALO;
    int gy = y + r - 1, gx = x0 + q - 1;
    float v = 0.f;
    if (gy >= 0 && gy < HH && gx >= 0 && gx < WW) {
      if (ch < CIN)      v = xt[(size_t)ch * (HH * WW) + gy * WW + gx];
      else if (s > 0)    v = hprev[(size_t)(ch - CIN) * (HH * WW) + gy * WW + gx];
    }
    lin[(r * HALO + q) * KDIM + ch] = (_Float16)v;
  }
  __syncthreads();

  const int lane = tid & 31;
  const int mt   = tid >> 5;      // wave id = M tile (gate channels mt*16..)
  const int n    = lane & 15;     // B/C/D column
  const int hl   = lane >> 4;     // half-wave select

  const v16h* apbase = (const v16h*)(apack + (size_t)dir * APACK_HALVES);

  v8f acc = {};
#pragma unroll
  for (int tap = 0; tap < 9; ++tap) {
    const int ky = tap / 3, kx = tap % 3;
    const _Float16* brow = &lin[(ky * HALO + n + kx) * KDIM + hl * 16];
#pragma unroll
    for (int kc = 0; kc < 2; ++kc) {
      v16h a = apbase[((mt * 9 + tap) * 2 + kc) * 32 + lane];
      v16h bm = *(const v16h*)(brow + kc * 32);
      acc = __builtin_amdgcn_wmma_f32_16x16x32_f16(
          /*neg_a=*/false, a, /*neg_b=*/false, bm,
          /*c_mod=*/(short)0, acc, /*reuse_a=*/false, /*reuse_b=*/false);
    }
  }

  // ---- bias + share gate tile through LDS ----
  const float* bias = dir ? bb : bf;
#pragma unroll
  for (int v = 0; v < 8; ++v) {
    int m = mt * 16 + v + hl * 8;   // C/D layout: lanes16-31 hold rows M+8
    zl[m * NPX + n] = acc[v] + bias[m];
  }
  __syncthreads();

  // ---- LSTM pointwise update: 512 (co,px) outputs, 2 per thread ----
  float* cdir = cstate + (size_t)dir * (B_ * CO * HH * WW);
#pragma unroll
  for (int k = 0; k < 2; ++k) {
    int o2 = tid + k * 256;
    int co = o2 >> 4, px = o2 & 15;
    float zi = zl[(co     ) * NPX + px];
    float zf = zl[(co + 32) * NPX + px];
    float zo = zl[(co + 64) * NPX + px];
    float zg = zl[(co + 96) * NPX + px];
    size_t cidx = ((size_t)(b * CO + co) * HH + y) * WW + x0 + px;
    float cp = (s > 0) ? cdir[cidx] : 0.f;
    float si = 1.f / (1.f + __expf(-zi));
    float sf = 1.f / (1.f + __expf(-zf));
    float so = 1.f / (1.f + __expf(-zo));
    float cn = sf * cp + si * tanhf(zg);
    float h  = so * tanhf(cn);
    cdir[cidx] = cn;
    out[((size_t)(s * B_ + b) * (2 * CO) + dir * CO + co) * (HH * WW)
        + y * WW + x0 + px] = h;
  }
}

extern "C" void kernel_launch(void* const* d_in, const int* in_sizes, int n_in,
                              void* d_out, int out_size, void* d_ws, size_t ws_size,
                              hipStream_t stream) {
  const float* x  = (const float*)d_in[0];
  const float* Wf = (const float*)d_in[1];
  const float* bf = (const float*)d_in[2];
  const float* Wb = (const float*)d_in[3];
  const float* bb = (const float*)d_in[4];
  float* out = (float*)d_out;

  // workspace: [0, 294912) packed f16 weights (both dirs); then 2x c-state f32
  _Float16* apack  = (_Float16*)d_ws;
  float*    cstate = (float*)((char*)d_ws + (size_t)2 * APACK_HALVES * sizeof(_Float16));

  repack_weights<<<(2 * APACK_HALVES + 255) / 256, 256, 0, stream>>>(Wf, Wb, apack);

  dim3 grid(B_ * HH * (WW / NPX), 2);  // 4096 pixel tiles x 2 directions
  for (int s = 0; s < S_; ++s)
    convlstm_step<<<grid, 256, 0, stream>>>(x, apack, bf, bb, out, cstate, s);
}
